// HierarchicalCodebook_90752658964799
// MI455X (gfx1250) — matrix operand order repaired and checked by reading, not compiled, not run on hardware
//
#include <hip/hip_runtime.h>

#define DDIM 320
#define N_CAT 20
#define N_TYPE 10
#define N_VAR 4
#define N_SPATIAL 20
#define N_TOTAL 1040          // 20 + 200 + 800 + 20
#define OFF_CAT 0
#define OFF_TYPE 20
#define OFF_VAR 220
#define OFF_SPAT 1020
#define M_TILE 16
#define N_TILES (N_TOTAL / 16)   // 65
#define K_STEPS (DDIM / 32)      // 10
#define SSTR 1044                // padded f32 stride for sim tile (bank-conflict free)
#define HSTR 328                 // padded bf16 stride for 16-row bf16 tiles
#define N_CS 40                  // cat + spatial rows staged in LDS

typedef __attribute__((ext_vector_type(16))) __bf16 v16bf;
typedef __attribute__((ext_vector_type(8)))  float  v8f;

// ---------------------------------------------------------------------------
// WMMA fold prologue:  Afold[n,:] = code_n @ W_level  (bf16), beff[n] = code_n.b
// Tiles: cat 0-1 | type 2-14 | var 15-64 | spat 65-66  => 67 blocks, 256 thr.
// ---------------------------------------------------------------------------
__global__ void __launch_bounds__(256)
hc_fold_kernel(const float* __restrict__ catC,
               const float* __restrict__ typC,
               const float* __restrict__ varC,
               const float* __restrict__ spatC,
               const float* __restrict__ Wc, const float* __restrict__ bc,
               const float* __restrict__ Wt, const float* __restrict__ bt,
               const float* __restrict__ Wv, const float* __restrict__ bv,
               const float* __restrict__ Ws, const float* __restrict__ bs,
               __bf16* __restrict__ Afold, float* __restrict__ beff) {
    __shared__ __bf16 cTile[M_TILE * HSTR];

    const int bid  = blockIdx.x;
    const int tid  = threadIdx.x;
    const int lane = tid & 31;
    const int wv   = tid >> 5;

    // level table
    const float* code; const float* W; const float* b;
    int rowAbs, nrows;
    if (bid < 2) {
        int t = bid;
        code = catC + (long)t * 16 * DDIM;  W = Wc; b = bc;
        rowAbs = OFF_CAT + t * 16;          nrows = min(16, N_CAT - t * 16);
    } else if (bid < 15) {
        int t = bid - 2;
        code = typC + (long)t * 16 * DDIM;  W = Wt; b = bt;
        rowAbs = OFF_TYPE + t * 16;         nrows = min(16, N_CAT * N_TYPE - t * 16);
    } else if (bid < 65) {
        int t = bid - 15;
        code = varC + (long)t * 16 * DDIM;  W = Wv; b = bv;
        rowAbs = OFF_VAR + t * 16;          nrows = 16;            // 800 = 50*16
    } else {
        int t = bid - 65;
        code = spatC + (long)t * 16 * DDIM; W = Ws; b = bs;
        rowAbs = OFF_SPAT + t * 16;         nrows = min(16, N_SPATIAL - t * 16);
    }

    // ---- load code tile -> bf16 LDS (zero-pad rows beyond level) ----
    for (int i = tid; i < M_TILE * DDIM; i += 256) {
        int r = i / DDIM, c = i - r * DDIM;
        float v = (r < nrows) ? code[(long)r * DDIM + c] : 0.f;
        cTile[r * HSTR + c] = (__bf16)v;
    }
    __syncthreads();

    // ---- hoist A fragments (code rows) ----
    const int r     = lane & 15;
    const int base0 = (lane >> 4) << 3;
    v16bf afrag[K_STEPS];
#pragma unroll
    for (int k = 0; k < K_STEPS; ++k) {
#pragma unroll
        for (int j = 0; j < 16; ++j) {
            int kk = k * 32 + base0 + j + ((j >= 8) ? 8 : 0);
            afrag[k][j] = cTile[r * HSTR + kk];
        }
    }

    // ---- GEMM: B[k][n] = W[k*DDIM + n]; 20 N-tiles over 8 waves ----
    const int koff  = (lane >> 4) << 4;
    const int rbase = (lane >> 4) << 3;
    for (int nt = wv; nt < DDIM / 16; nt += 8) {
        const int nBase = nt * 16;
        const int col   = nBase + (lane & 15);
        v8f acc = {};
#pragma unroll
        for (int k = 0; k < K_STEPS; ++k) {
            v16bf bfrag;
#pragma unroll
            for (int j = 0; j < 16; ++j)
                bfrag[j] = (__bf16)W[(long)(k * 32 + ((lane >> 4) << 3) + j + ((j >= 8) ? 8 : 0)) * DDIM + col];
            acc = __builtin_amdgcn_wmma_f32_16x16x32_bf16(
                      false, afrag[k], false, bfrag, (short)0, acc, false, false);
        }
#pragma unroll
        for (int g = 0; g < 8; ++g) {
            int m = g + rbase;
            if (m < nrows)
                Afold[(long)(rowAbs + m) * DDIM + col] = (__bf16)acc[g];
        }
    }

    // ---- beff: 8 waves x 2 rows, full-wave reduction ----
    for (int tt = 0; tt < 2; ++tt) {
        int m = wv * 2 + tt;
        if (m < nrows) {
            const float* cr = code + (long)m * DDIM;
            float p = 0.f;
#pragma unroll
            for (int i = 0; i < DDIM / 32; ++i)
                p = fmaf(cr[lane + 32 * i], b[lane + 32 * i], p);
            for (int off = 16; off > 0; off >>= 1)
                p += __shfl_xor(p, off, 32);
            if (lane == 0) beff[rowAbs + m] = p;
        }
    }
}

// ---------------------------------------------------------------------------
// Main fused kernel: 16 tokens per block, 256 threads (8 waves).
// ---------------------------------------------------------------------------
__global__ void __launch_bounds__(256)
hc_main_kernel(const float* __restrict__ x,
               const float* __restrict__ catC,
               const float* __restrict__ typC,
               const float* __restrict__ varC,
               const float* __restrict__ spatC,
               const float* __restrict__ log_tau,
               const __bf16* __restrict__ Afold,
               const float* __restrict__ beff,
               float* __restrict__ out,
               long T) {
    extern __shared__ char smem_raw[];
    __bf16* hTile  = (__bf16*)smem_raw;                                  // 16*328 bf16
    float*  S      = (float*)(smem_raw + (size_t)M_TILE * HSTR * 2);     // 16*1044 f32
    float*  codeCS = (float*)(smem_raw + (size_t)M_TILE * HSTR * 2
                                       + (size_t)M_TILE * SSTR * 4);     // 40*320 f32
    int*    selCat  = (int*)((char*)codeCS + (size_t)N_CS * DDIM * 4);
    int*    selType = selCat + M_TILE;

    const int tid  = threadIdx.x;
    const int lane = tid & 31;
    const int wv   = tid >> 5;
    const long tokBase = (long)blockIdx.x * M_TILE;

    // ---- cooperative loads: x tile -> bf16, cat+spatial codes -> LDS ----
    {
        const float* xs = x + tokBase * DDIM;
        const long   lim = (T - tokBase) * DDIM;
        for (int i = tid; i < M_TILE * DDIM; i += 256) {
            int r = i / DDIM, c = i - r * DDIM;
            float v = (i < lim) ? xs[i] : 0.f;
            hTile[r * HSTR + c] = (__bf16)v;
        }
        for (int i = tid; i < N_CS * DDIM; i += 256)
            codeCS[i] = (i < N_CAT * DDIM) ? catC[i] : spatC[i - N_CAT * DDIM];
    }
    __syncthreads();

    float tau = __expf(log_tau[0]) + 0.1f;
    tau = fminf(fmaxf(tau, 0.1f), 2.0f);
    const float invTau = 1.0f / tau;

    // ---- hoist all 10 A-matrix (token) fragments into VGPRs ----
    const int r     = lane & 15;
    const int base0 = (lane >> 4) << 4;
    v16bf afrag[K_STEPS];
#pragma unroll
    for (int k = 0; k < K_STEPS; ++k) {
#pragma unroll
        for (int j = 0; j < 16; ++j) {
            int kk = k * 32 + base0 + j;
            afrag[k][j] = hTile[r * HSTR + kk];
        }
    }

    // ---- GEMM over the 65 N-tiles (wave-strided) ----
    const int koff = (lane >> 4) << 4;
    for (int nt = wv; nt < N_TILES; nt += 8) {
        const int nBase = nt * 16;
        const int nRow  = nBase + (lane & 15);
        const __bf16* bp = Afold + (long)nRow * DDIM + koff;
        v8f acc = {};
#pragma unroll
        for (int k = 0; k < K_STEPS; ++k) {
            v16bf bfrag = *(const v16bf*)(bp + k * 32);
            acc = __builtin_amdgcn_wmma_f32_16x16x32_bf16(
                      false, afrag[k], false, bfrag, (short)0, acc, false, false);
        }
        const int   col   = nBase + (lane & 15);
        const float be    = beff[col];
        const int   rbase = (lane >> 4) << 3;
#pragma unroll
        for (int g = 0; g < 8; ++g)
            S[(g + rbase) * SSTR + col] = (acc[g] + be) * invTau;
    }
    __syncthreads();

    // ---- per-token epilogue ----
    for (int tt = 0; tt < 2; ++tt) {
        const int t = wv * 2 + tt;
        float* Srow = S + t * SSTR;

        if (lane == 0) {
            float m = -3.4e38f; int am = 0;
            for (int n = 0; n < N_CAT; ++n) { float v = Srow[OFF_CAT + n]; if (v > m) { m = v; am = n; } }
            float sum = 0.f;
            for (int n = 0; n < N_CAT; ++n) { float e = __expf(Srow[OFF_CAT + n] - m); Srow[OFF_CAT + n] = e; sum += e; }
            float inv = 1.f / sum;
            for (int n = 0; n < N_CAT; ++n) Srow[OFF_CAT + n] *= inv;
            selCat[t] = am;

            const int tb = OFF_TYPE + am * N_TYPE;
            m = -3.4e38f; int at = 0;
            for (int n = 0; n < N_TYPE; ++n) { float v = Srow[tb + n]; if (v > m) { m = v; at = n; } }
            sum = 0.f;
            for (int n = 0; n < N_TYPE; ++n) { float e = __expf(Srow[tb + n] - m); Srow[tb + n] = e; sum += e; }
            inv = 1.f / sum;
            for (int n = 0; n < N_TYPE; ++n) Srow[tb + n] *= inv;
            const int tflat = am * N_TYPE + at;
            selType[t] = tflat;

            const int vb = OFF_VAR + tflat * N_VAR;
            m = -3.4e38f;
            for (int n = 0; n < N_VAR; ++n) m = fmaxf(m, Srow[vb + n]);
            sum = 0.f;
            for (int n = 0; n < N_VAR; ++n) { float e = __expf(Srow[vb + n] - m); Srow[vb + n] = e; sum += e; }
            inv = 1.f / sum;
            for (int n = 0; n < N_VAR; ++n) Srow[vb + n] *= inv;

            m = -3.4e38f;
            for (int n = 0; n < N_SPATIAL; ++n) m = fmaxf(m, Srow[OFF_SPAT + n]);
            sum = 0.f;
            for (int n = 0; n < N_SPATIAL; ++n) { float e = __expf(Srow[OFF_SPAT + n] - m); Srow[OFF_SPAT + n] = e; sum += e; }
            inv = 1.f / sum;
            for (int n = 0; n < N_SPATIAL; ++n) Srow[OFF_SPAT + n] *= inv;
        }
        __syncthreads();

        float z[DDIM / 32];
#pragma unroll
        for (int i = 0; i < DDIM / 32; ++i) z[i] = 0.f;

        for (int n = 0; n < N_CAT; ++n) {                 // staged in LDS
            const float p = Srow[OFF_CAT + n];
            const float* c = codeCS + n * DDIM;
#pragma unroll
            for (int i = 0; i < DDIM / 32; ++i) z[i] = fmaf(p, c[lane + 32 * i], z[i]);
        }
        const int cat = selCat[t];
        for (int n = 0; n < N_TYPE; ++n) {                // data-dependent: L2
            const float p = Srow[OFF_TYPE + cat * N_TYPE + n];
            const float* c = typC + (long)(cat * N_TYPE + n) * DDIM;
#pragma unroll
            for (int i = 0; i < DDIM / 32; ++i) z[i] = fmaf(p, c[lane + 32 * i], z[i]);
        }
        const int tf = selType[t];
        for (int n = 0; n < N_VAR; ++n) {                 // data-dependent: L2
            const float p = Srow[OFF_VAR + tf * N_VAR + n];
            const float* c = varC + (long)(tf * N_VAR + n) * DDIM;
#pragma unroll
            for (int i = 0; i < DDIM / 32; ++i) z[i] = fmaf(p, c[lane + 32 * i], z[i]);
        }
        for (int n = 0; n < N_SPATIAL; ++n) {             // staged in LDS
            const float p = Srow[OFF_SPAT + n];
            const float* c = codeCS + (N_CAT + n) * DDIM;
#pragma unroll
            for (int i = 0; i < DDIM / 32; ++i) z[i] = fmaf(p, c[lane + 32 * i], z[i]);
        }

        const long tok = tokBase + t;
        if (tok < T) {
            float* o = out + tok * DDIM;
#pragma unroll
            for (int i = 0; i < DDIM / 32; ++i) o[lane + 32 * i] = z[i];
        }
        __syncthreads();
    }
}

// ---------------------------------------------------------------------------
extern "C" void kernel_launch(void* const* d_in, const int* in_sizes, int n_in,
                              void* d_out, int out_size, void* d_ws, size_t ws_size,
                              hipStream_t stream) {
    const float* x     = (const float*)d_in[0];
    const float* catC  = (const float*)d_in[1];
    const float* typC  = (const float*)d_in[2];
    const float* varC  = (const float*)d_in[3];
    const float* spatC = (const float*)d_in[4];
    const float* ltau  = (const float*)d_in[5];
    const float* Wc    = (const float*)d_in[6];
    const float* bc    = (const float*)d_in[7];
    const float* Wt    = (const float*)d_in[8];
    const float* bt    = (const float*)d_in[9];
    const float* Wv    = (const float*)d_in[10];
    const float* bv    = (const float*)d_in[11];
    const float* Ws    = (const float*)d_in[12];
    const float* bs    = (const float*)d_in[13];
    float* out = (float*)d_out;

    const long T = (long)in_sizes[0] / DDIM;   // 32768 tokens

    __bf16* Afold = (__bf16*)d_ws;                                   // 1040*320 bf16
    float*  beff  = (float*)((char*)d_ws + (size_t)N_TOTAL * DDIM * 2);

    hc_fold_kernel<<<67, 256, 0, stream>>>(
        catC, typC, varC, spatC, Wc, bc, Wt, bt, Wv, bv, Ws, bs, Afold, beff);

    const size_t smem = (size_t)M_TILE * HSTR * 2
                      + (size_t)M_TILE * SSTR * 4
                      + (size_t)N_CS * DDIM * 4
                      + 2 * M_TILE * sizeof(int);
    const long nBlocks = (T + M_TILE - 1) / M_TILE;
    hc_main_kernel<<<dim3((unsigned)nBlocks), dim3(256), smem, stream>>>(
        x, catC, typC, varC, spatC, ltau, Afold, beff, out, T);
}
